// RegionOutputLayer_73641509257811
// MI455X (gfx1250) — matrix operand order repaired and checked by reading, not compiled, not run on hardware
//
#include <hip/hip_runtime.h>
#include <hip/hip_bf16.h>

// ---------------------------------------------------------------------------
// RegionOutputLayer for gfx1250 (CDNA5, wave32, WMMA)
//   pre  : NCHW f32 -> zero-halo-padded NHWC f16 input (42x42 planes);
//          conv1_w -> tap-major f16; conv2_w -> zero-padded f16
//   conv1: 3x3 SAME, 512->512, implicit GEMM  M=25600 N=512 K=4608 (f16 WMMA)
//          tap-outer K order; halo padding -> unconditional A staging
//   conv2: 1x1, 512->425, GEMM               M=25600 N=425 K=512  (f16 WMMA)
//   post : per-anchor box decode + sigmoid + softmax
// Both GEMMs: 128x128 block tile, 8 wave32 waves (4Mx2N), K step 32,
// double-buffered LDS, reg-prefetch pipeline, one LDS-only barrier per step
// (s_wait_dscnt + s_barrier_signal/-wait; no loadcnt fence at the barrier).
// ---------------------------------------------------------------------------

typedef __attribute__((ext_vector_type(16))) _Float16 v16h;
typedef __attribute__((ext_vector_type(8)))  _Float16 v8h;
typedef __attribute__((ext_vector_type(8)))  float    v8f;

#define BATCH   16
#define CIN     512
#define HH      40
#define WW      40
#define HW      (HH * WW)             // 1600
#define PADW    42                    // halo-padded plane width/height
#define PADHW   (PADW * PADW)         // 1764
#define CMID    512
#define COUT    425
#define NANCH   5
#define NCLS    80
#define MTOT    (BATCH * HH * WW)     // 25600
#define LDSTR   40                    // LDS row stride in halves (80B, 16B aligned)
#define NT1     144                   // conv1 K steps: 9 taps * 16 ci-chunks
#define NT2     16                    // conv2 K steps: 512 / 32

__device__ __forceinline__ float sigmoidf_(float x) {
  return 1.0f / (1.0f + __expf(-x));
}

// LDS-only barrier: waits LDS ops, NOT outstanding global loads, so the
// register-prefetch of the next tile keeps flying across the barrier.
__device__ __forceinline__ void lds_barrier() {
  asm volatile("s_wait_dscnt 0x0\n\t"
               "s_barrier_signal -1\n\t"
               "s_barrier_wait -1" ::: "memory");
}

__device__ __forceinline__ v16h cat8(v8h lo, v8h hi) {
  return __builtin_shufflevector(lo, hi, 0, 1, 2, 3, 4, 5, 6, 7,
                                         8, 9, 10, 11, 12, 13, 14, 15);
}

// A fragment (16x32 f16): M = lane&15; lanes<16 hold K {0..7,16..23},
// lanes>=16 hold K {8..15,24..31}. Two contiguous 16B runs per lane.
__device__ __forceinline__ v16h load_a_frag(const _Float16* base, int lane) {
  const int kb = (lane < 16) ? 0 : 8;
  const _Float16* row = base + (lane & 15) * LDSTR;
  v8h lo = *(const v8h*)(row + kb);
  v8h hi = *(const v8h*)(row + 16 + kb);
  return cat8(lo, hi);
}

// B fragment (32x16 f16) from K-transposed LDS tile Bt[n][k]:
// N = lane&15; lanes<16 hold K {0..7,16..23}, lanes>=16 hold K {8..15,24..31}.
__device__ __forceinline__ v16h load_b_frag(const _Float16* base, int lane) {
  const int kb = (lane < 16) ? 0 : 8;
  const _Float16* row = base + (lane & 15) * LDSTR + kb;
  v8h lo = *(const v8h*)(row);
  v8h hi = *(const v8h*)(row + 16);
  return cat8(lo, hi);
}

// ---------------------------------------------------------------------------
// Pre-transform kernels (one-time, bandwidth-trivial)
// ---------------------------------------------------------------------------

// Zero-fill the padded NHWC buffer (covers the halo; interior overwritten).
__global__ __launch_bounds__(256)
void zero_pad_kernel(_Float16* __restrict__ buf, int n16) {
  const int i = blockIdx.x * blockDim.x + threadIdx.x;
  if (i >= n16) return;
  const v8h z = {};
  *(v8h*)(buf + (size_t)i * 16)     = z;
  *(v8h*)(buf + (size_t)i * 16 + 8) = z;
}

// NCHW f32 -> padded NHWC f16 via 32x32 LDS tile transpose; coalesced both
// sides. grid = (HW/32, CIN/32, BATCH), block = 256
__global__ __launch_bounds__(256)
void nchw_to_nhwc_kernel(const float* __restrict__ in,
                         _Float16* __restrict__ out) {
  __shared__ float tile[32][33];
  const int b   = blockIdx.z;
  const int hw0 = blockIdx.x * 32;
  const int ci0 = blockIdx.y * 32;
  const int tx  = threadIdx.x & 31;
  const int ty  = threadIdx.x >> 5;   // 0..7

  const float* src = in + ((size_t)b * CIN + ci0) * HW + hw0;
#pragma unroll
  for (int r = 0; r < 4; ++r) {
    const int ci = ty + r * 8;
    tile[ci][tx] = src[(size_t)ci * HW + tx];   // coalesced along hw
  }
  __syncthreads();
  _Float16* dstb = out + (size_t)b * PADHW * CIN;
#pragma unroll
  for (int r = 0; r < 4; ++r) {
    const int hw = hw0 + ty + r * 8;
    const int y  = hw / WW;
    const int x  = hw - y * WW;
    // interior of padded plane: (y+1, x+1)
    dstb[((size_t)(y + 1) * PADW + (x + 1)) * CIN + ci0 + tx] =
        (_Float16)tile[tx][ty + r * 8];         // coalesced along ci
  }
}

// w1t[tap][n][ci] = (f16) w1[n][ci][tap]   (OIHW -> tap-major, coalesced writes)
__global__ __launch_bounds__(256)
void w1_transform_kernel(const float* __restrict__ w1,
                         _Float16* __restrict__ w1t) {
  const int i = blockIdx.x * blockDim.x + threadIdx.x;
  if (i >= 9 * CMID * CIN) return;
  const int ci  = i & (CIN - 1);
  const int n   = (i >> 9) & (CMID - 1);
  const int tap = i >> 18;
  w1t[i] = (_Float16)w1[((size_t)n * CIN + ci) * 9 + tap];
}

// w2h[n][k] f16, zero-padded to 512 output rows
__global__ __launch_bounds__(256)
void w2_transform_kernel(const float* __restrict__ w2,
                         _Float16* __restrict__ w2h) {
  const int i = blockIdx.x * blockDim.x + threadIdx.x;
  if (i >= 512 * CMID) return;
  const int k = i & (CMID - 1);
  const int n = i >> 9;
  const float v = (n < COUT) ? w2[(size_t)n * CMID + k] : 0.0f;
  w2h[i] = (_Float16)v;
}

// ---------------------------------------------------------------------------
// Kernel 1: 3x3 conv (implicit GEMM, tap-outer, padded NHWC) + BN + SiLU
// ---------------------------------------------------------------------------
__global__ __launch_bounds__(256)
void conv1_bn_silu_kernel(const _Float16* __restrict__ in_pad,
                          const _Float16* __restrict__ w1t,
                          const float* __restrict__ gamma,
                          const float* __restrict__ beta,
                          const float* __restrict__ mean,
                          const float* __restrict__ var,
                          _Float16* __restrict__ h_out) {
  __shared__ alignas(16) _Float16 sA[2][128 * LDSTR];
  __shared__ alignas(16) _Float16 sB[2][128 * LDSTR];

  const int tid    = threadIdx.x;
  const int lane   = tid & 31;
  const int wave   = tid >> 5;
  const int wave_m = wave >> 1;    // 0..3 -> 32 rows
  const int wave_n = wave & 1;     // 0..1 -> 64 cols

  const int n_blk = blockIdx.x & 3;
  const int m_blk = blockIdx.x >> 2;
  const int m0 = m_blk * 128;
  const int n0 = n_blk * 128;

  // Staging: 256 threads cover 128(row) x 32(k), 16 k-contiguous elems each.
  const int st_mn = tid & 127;
  const int st_kh = (tid >> 7) << 4;       // 0 or 16

  // Spatial decode for the A row this thread stages (fixed for whole kernel).
  const int mg = m0 + st_mn;
  const int bb = mg / HW;
  const int rm = mg - bb * HW;
  const int yy = rm / WW;
  const int xx = rm - yy * WW;

  // Thread-fixed base pointers; per-tile offsets are scalar/tiny.
  const _Float16* pA_base =
      in_pad + ((size_t)bb * PADHW + (size_t)yy * PADW + xx) * CIN + st_kh;
  const _Float16* pB_base = w1t + (size_t)(n0 + st_mn) * CIN + st_kh;

  v8h pa0, pa1, pb0, pb1;
  auto prefetch = [&](int t) {
    const int tap = t >> 4;          // 0..8 (uniform -> scalar)
    const int cs  = t & 15;
    const int ky  = tap / 3;
    const int kx  = tap - ky * 3;
    // halo-padded: (yy+ky, xx+kx) is always in-bounds -> no guard
    const _Float16* p = pA_base + ((size_t)(ky * PADW + kx)) * CIN + (cs << 5);
    pa0 = *(const v8h*)(p);
    pa1 = *(const v8h*)(p + 8);
    const _Float16* q = pB_base + (size_t)tap * (CMID * CIN) + (cs << 5);
    pb0 = *(const v8h*)(q);
    pb1 = *(const v8h*)(q + 8);
  };

  v8f acc[2][4];
  {
    v8f z = {};
#pragma unroll
    for (int i = 0; i < 2; ++i)
#pragma unroll
      for (int j = 0; j < 4; ++j) acc[i][j] = z;
  }

  // Pipeline prologue: stage tile 0 into buffer 0.
  prefetch(0);
  *(v8h*)(&sA[0][st_mn * LDSTR + st_kh])     = pa0;
  *(v8h*)(&sA[0][st_mn * LDSTR + st_kh + 8]) = pa1;
  *(v8h*)(&sB[0][st_mn * LDSTR + st_kh])     = pb0;
  *(v8h*)(&sB[0][st_mn * LDSTR + st_kh + 8]) = pb1;

  for (int t = 0; t < NT1; ++t) {
    const int cur = t & 1;
    if (t + 1 < NT1) prefetch(t + 1);   // global loads overlap WMMA below
    lds_barrier();                      // buf[cur] ready; prior reads done

    v16h afr[2], bfr[4];
#pragma unroll
    for (int u = 0; u < 2; ++u)
      afr[u] = load_a_frag(&sA[cur][(wave_m * 32 + u * 16) * LDSTR], lane);
#pragma unroll
    for (int u = 0; u < 4; ++u)
      bfr[u] = load_b_frag(&sB[cur][(wave_n * 64 + u * 16) * LDSTR], lane);

#pragma unroll
    for (int i = 0; i < 2; ++i)
#pragma unroll
      for (int j = 0; j < 4; ++j)
        acc[i][j] = __builtin_amdgcn_wmma_f32_16x16x32_f16(
            false, afr[i], false, bfr[j], (short)0, acc[i][j], false, false);

    if (t + 1 < NT1) {
      const int nxt = cur ^ 1;
      *(v8h*)(&sA[nxt][st_mn * LDSTR + st_kh])     = pa0;
      *(v8h*)(&sA[nxt][st_mn * LDSTR + st_kh + 8]) = pa1;
      *(v8h*)(&sB[nxt][st_mn * LDSTR + st_kh])     = pb0;
      *(v8h*)(&sB[nxt][st_mn * LDSTR + st_kh + 8]) = pb1;
    }
  }

  // Fused BN (eval stats) + SiLU epilogue, f16 h[m][cmid]
  const int lm = (lane < 16) ? 0 : 8;
  const int ln = lane & 15;
#pragma unroll
  for (int j = 0; j < 4; ++j) {
    const int ng = n0 + wave_n * 64 + j * 16 + ln;
    const float s  = gamma[ng] * __frsqrt_rn(var[ng] + 1e-5f);
    const float mu = mean[ng];
    const float be = beta[ng];
#pragma unroll
    for (int i = 0; i < 2; ++i) {
#pragma unroll
      for (int r = 0; r < 8; ++r) {
        const int mrow = m0 + wave_m * 32 + i * 16 + lm + r;
        float v = (acc[i][j][r] - mu) * s + be;
        v = v * sigmoidf_(v);  // SiLU
        h_out[(size_t)mrow * CMID + ng] = (_Float16)v;
      }
    }
  }
}

// ---------------------------------------------------------------------------
// Kernel 2: 1x1 conv GEMM  det[m][425] = h[m][512] * w2^T + bias
// ---------------------------------------------------------------------------
__global__ __launch_bounds__(256)
void conv2_kernel(const _Float16* __restrict__ h,
                  const _Float16* __restrict__ w2h,
                  const float* __restrict__ bias,
                  float* __restrict__ det) {
  __shared__ alignas(16) _Float16 sA[2][128 * LDSTR];
  __shared__ alignas(16) _Float16 sB[2][128 * LDSTR];

  const int tid    = threadIdx.x;
  const int lane   = tid & 31;
  const int wave   = tid >> 5;
  const int wave_m = wave >> 1;
  const int wave_n = wave & 1;

  const int n_blk = blockIdx.x & 3;
  const int m_blk = blockIdx.x >> 2;
  const int m0 = m_blk * 128;
  const int n0 = n_blk * 128;

  const int st_mn = tid & 127;
  const int st_kh = (tid >> 7) << 4;

  const _Float16* arow0 = h + (size_t)(m0 + st_mn) * CMID + st_kh;
  const _Float16* brow0 = w2h + (size_t)(n0 + st_mn) * CMID + st_kh;

  v8h pa0, pa1, pb0, pb1;
  auto prefetch = [&](int t) {
    const _Float16* a = arow0 + (t << 5);
    const _Float16* b = brow0 + (t << 5);
    pa0 = *(const v8h*)(a);
    pa1 = *(const v8h*)(a + 8);
    pb0 = *(const v8h*)(b);
    pb1 = *(const v8h*)(b + 8);
  };

  v8f acc[2][4];
  {
    v8f z = {};
#pragma unroll
    for (int i = 0; i < 2; ++i)
#pragma unroll
      for (int j = 0; j < 4; ++j) acc[i][j] = z;
  }

  prefetch(0);
  *(v8h*)(&sA[0][st_mn * LDSTR + st_kh])     = pa0;
  *(v8h*)(&sA[0][st_mn * LDSTR + st_kh + 8]) = pa1;
  *(v8h*)(&sB[0][st_mn * LDSTR + st_kh])     = pb0;
  *(v8h*)(&sB[0][st_mn * LDSTR + st_kh + 8]) = pb1;

  for (int t = 0; t < NT2; ++t) {
    const int cur = t & 1;
    if (t + 1 < NT2) prefetch(t + 1);
    lds_barrier();

    v16h afr[2], bfr[4];
#pragma unroll
    for (int u = 0; u < 2; ++u)
      afr[u] = load_a_frag(&sA[cur][(wave_m * 32 + u * 16) * LDSTR], lane);
#pragma unroll
    for (int u = 0; u < 4; ++u)
      bfr[u] = load_b_frag(&sB[cur][(wave_n * 64 + u * 16) * LDSTR], lane);

#pragma unroll
    for (int i = 0; i < 2; ++i)
#pragma unroll
      for (int j = 0; j < 4; ++j)
        acc[i][j] = __builtin_amdgcn_wmma_f32_16x16x32_f16(
            false, afr[i], false, bfr[j], (short)0, acc[i][j], false, false);

    if (t + 1 < NT2) {
      const int nxt = cur ^ 1;
      *(v8h*)(&sA[nxt][st_mn * LDSTR + st_kh])     = pa0;
      *(v8h*)(&sA[nxt][st_mn * LDSTR + st_kh + 8]) = pa1;
      *(v8h*)(&sB[nxt][st_mn * LDSTR + st_kh])     = pb0;
      *(v8h*)(&sB[nxt][st_mn * LDSTR + st_kh + 8]) = pb1;
    }
  }

  const int lm = (lane < 16) ? 0 : 8;
  const int ln = lane & 15;
#pragma unroll
  for (int j = 0; j < 4; ++j) {
    const int ng = n0 + wave_n * 64 + j * 16 + ln;
    if (ng >= COUT) continue;
    const float bv = bias[ng];
#pragma unroll
    for (int i = 0; i < 2; ++i) {
#pragma unroll
      for (int r = 0; r < 8; ++r) {
        const int mrow = m0 + wave_m * 32 + i * 16 + lm + r;
        det[(size_t)mrow * COUT + ng] = acc[i][j][r] + bv;
      }
    }
  }
}

// ---------------------------------------------------------------------------
// Kernel 3: box decode + objectness + softmax. One thread per (cell, anchor).
// out = [boxes (B,H,W,A,4) | obj (B,H,W,A) | classes (B,H,W,A,NC)]
// ---------------------------------------------------------------------------
__global__ __launch_bounds__(256)
void postprocess_kernel(const float* __restrict__ det,
                        const float* __restrict__ anchors,
                        float* __restrict__ out) {
  const int idx = blockIdx.x * blockDim.x + threadIdx.x;  // m*5 + a
  if (idx >= MTOT * NANCH) return;
  const int a = idx % NANCH;
  const int m = idx / NANCH;
  const int x = m % WW;
  const int y = (m / WW) % HH;

  const float* d = det + (size_t)m * COUT + a * (5 + NCLS);

  float* boxes = out;
  float* obj   = out + (size_t)MTOT * NANCH * 4;
  float* cls   = obj + (size_t)MTOT * NANCH;

  boxes[(size_t)idx * 4 + 0] = ((float)x + sigmoidf_(d[0])) * (1.0f / WW);
  boxes[(size_t)idx * 4 + 1] = ((float)y + sigmoidf_(d[1])) * (1.0f / HH);
  boxes[(size_t)idx * 4 + 2] = anchors[a * 2 + 0] * __expf(d[2]);
  boxes[(size_t)idx * 4 + 3] = anchors[a * 2 + 1] * __expf(d[3]);
  obj[idx] = sigmoidf_(d[4]);

  float mx = d[5];
  for (int c = 1; c < NCLS; ++c) mx = fmaxf(mx, d[5 + c]);
  float sum = 0.0f;
  for (int c = 0; c < NCLS; ++c) sum += __expf(d[5 + c] - mx);
  const float inv = 1.0f / sum;
  float* co = cls + (size_t)idx * NCLS;
  for (int c = 0; c < NCLS; ++c) co[c] = __expf(d[5 + c] - mx) * inv;
}

// ---------------------------------------------------------------------------
extern "C" void kernel_launch(void* const* d_in, const int* in_sizes, int n_in,
                              void* d_out, int out_size, void* d_ws, size_t ws_size,
                              hipStream_t stream) {
  const float* in      = (const float*)d_in[0];
  const float* conv1_w = (const float*)d_in[1];
  const float* gamma   = (const float*)d_in[2];
  const float* beta    = (const float*)d_in[3];
  const float* mean    = (const float*)d_in[4];
  const float* var     = (const float*)d_in[5];
  const float* conv2_w = (const float*)d_in[6];
  const float* conv2_b = (const float*)d_in[7];
  const float* anchors = (const float*)d_in[8];

  // Workspace layout (16B-aligned sections)
  char* ws = (char*)d_ws;
  _Float16* h_ws   = (_Float16*)(ws);                          // 26,214,400 B
  float*    det    = (float*)(ws + (size_t)MTOT * CMID * 2);   // 43,520,000 B
  _Float16* in_pad = (_Float16*)(ws + 69734400ull);            // 28,901,376 B
  _Float16* w1t    = (_Float16*)(ws + 98635776ull);            //  4,718,592 B
  _Float16* w2h    = (_Float16*)(ws + 103354368ull);           //    524,288 B

  const dim3 blk(256);

  const int npad16 = (BATCH * PADHW * CIN) / 16;  // 903,168
  zero_pad_kernel<<<(npad16 + 255) / 256, blk, 0, stream>>>(in_pad, npad16);
  nchw_to_nhwc_kernel<<<dim3(HW / 32, CIN / 32, BATCH), blk, 0, stream>>>(
      in, in_pad);
  w1_transform_kernel<<<(9 * CMID * CIN + 255) / 256, blk, 0, stream>>>(conv1_w, w1t);
  w2_transform_kernel<<<(512 * CMID + 255) / 256, blk, 0, stream>>>(conv2_w, w2h);

  const dim3 grid_gemm((MTOT / 128) * 4);  // 200 M-tiles x 4 N-tiles
  conv1_bn_silu_kernel<<<grid_gemm, blk, 0, stream>>>(
      in_pad, w1t, gamma, beta, mean, var, h_ws);
  conv2_kernel<<<grid_gemm, blk, 0, stream>>>(h_ws, w2h, conv2_b, det);
  postprocess_kernel<<<(MTOT * NANCH + 255) / 256, blk, 0, stream>>>(
      det, anchors, (float*)d_out);
}
